// WaveNet_56444460204090
// MI455X (gfx1250) — compile-verified, matrix operand order checked by reading, not run on hardware
//
#include <hip/hip_runtime.h>
#include <math.h>

// ---------------- types ----------------
typedef __attribute__((ext_vector_type(16))) __bf16 bf16x16;
typedef __attribute__((ext_vector_type(8)))  float  f32x8;

#define TT     18308      // full time length
#define SP     18312      // padded time rows (mult of 8)
#define TOUT   16262      // output time length (TT - 2046)
#define TOUTP  16264      // padded
#define NLAYER 20

__device__ __forceinline__ unsigned short f2bf(float f) {
  unsigned int u = __float_as_uint(f);
  u += 0x7FFFu + ((u >> 16) & 1u);          // round-to-nearest-even
  return (unsigned short)(u >> 16);
}
__device__ __forceinline__ float bf2f(unsigned short h) {
  return __uint_as_float(((unsigned int)h) << 16);
}

#define WMMA_BF16(A, B, C) \
  __builtin_amdgcn_wmma_f32_16x16x32_bf16(false, (A), false, (B), (short)0, (C), false, false)

// gfx1250 async global->LDS copy (bypasses VGPRs, tracked by ASYNCcnt)
__device__ __forceinline__ void async_ld_b128(void* lds, const void* gsrc) {
  unsigned l = (unsigned)(size_t)lds;       // LDS byte address = low 32 bits of generic ptr
  asm volatile("global_load_async_to_lds_b128 %0, %1, off"
               :: "v"(l), "v"(gsrc) : "memory");
}
__device__ __forceinline__ void wait_async0() {
  asm volatile("s_wait_asynccnt 0" ::: "memory");
}

// ---------------- weight prepack ----------------
// Wf/Wg: per layer 256 x 1024 bf16 = [sig tap0 (384) | sig tap1 (384) | psig (256)]
__global__ void k_pack_fg(const float* __restrict__ sig_w, const float* __restrict__ gate_w,
                          const float* __restrict__ psig_w, const float* __restrict__ pgate_w,
                          unsigned short* __restrict__ Wf, unsigned short* __restrict__ Wg) {
  long long idx = (long long)blockIdx.x * blockDim.x + threadIdx.x;
  if (idx >= 20ll * 256 * 1024) return;
  int k = (int)(idx & 1023);
  long long id = idx >> 10;            // i*256 + d
  float f, g;
  if (k < 384)      { f = sig_w[(id*384 + k)*2 + 0];        g = gate_w[(id*384 + k)*2 + 0]; }
  else if (k < 768) { f = sig_w[(id*384 + (k-384))*2 + 1];  g = gate_w[(id*384 + (k-384))*2 + 1]; }
  else              { f = psig_w[id*256 + (k-768)];         g = pgate_w[id*256 + (k-768)]; }
  Wf[idx] = f2bf(f); Wg[idx] = f2bf(g);
}

__global__ void k_tob(const float* __restrict__ a, unsigned short* __restrict__ b, long long n) {
  long long i = (long long)blockIdx.x * blockDim.x + threadIdx.x;
  if (i < n) b[i] = f2bf(a[i]);
}

__global__ void k_zero(float* __restrict__ p, long long n) {
  long long i = (long long)blockIdx.x * blockDim.x + threadIdx.x;
  if (i < n) p[i] = 0.f;
}

// ---------------- conditioning pipeline (small, scalar f32, channel-major scratch) ----------------
__global__ void k_lc0(const float* __restrict__ lcs, const int* __restrict__ jit,
                      const float* __restrict__ w, const float* __restrict__ b,
                      float* __restrict__ y) {
  int n = blockIdx.z;
  int idx = blockIdx.x * blockDim.x + threadIdx.x;
  if (idx >= 128 * 62) return;
  int o = idx / 62, s = idx % 62;
  float acc = b[o];
  for (int k = 0; k < 3; ++k) {
    int col = jit[n*64 + s + k];
    const float* xr = lcs + (size_t)n*64*64 + col;
    const float* wr = w + ((size_t)o*64)*3 + k;
    for (int i = 0; i < 64; ++i) acc += wr[(size_t)i*3] * xr[(size_t)i*64];
  }
  y[((size_t)(n*128) + o)*62 + s] = acc;
}

// transposed conv: y[n,o,t] = b[o] + sum_k{p=t+k-(s-1); p%s==0} sum_i w[i,o,K-1-k]*x[n,i,p/s]
__global__ void k_tconv(const float* __restrict__ x, const float* __restrict__ w,
                        const float* __restrict__ bias, float* __restrict__ y,
                        int C, int Lin, int Lout, int Kw, int s) {
  int n = blockIdx.z;
  long long idx = (long long)blockIdx.x * blockDim.x + threadIdx.x;
  if (idx >= (long long)C * Lout) return;
  int o = (int)(idx / Lout), t = (int)(idx % Lout);
  float acc = bias[o];
  for (int k = 0; k < Kw; ++k) {
    int p = t + k - (s - 1);
    if (p < 0 || (p % s)) continue;
    int q = p / s; if (q >= Lin) continue;
    const float* wr = w + (size_t)o*Kw + (Kw - 1 - k);
    const float* xr = x + ((size_t)n*C)*Lin + q;
    float a = 0.f;
    for (int i = 0; i < C; ++i) a += wr[(size_t)i*C*Kw] * xr[(size_t)i*Lin];
    acc += a;
  }
  y[((size_t)(n*C) + o)*Lout + t] = acc;
}

// cond bf16, time-major [n][t][256]: ch 0..127 = lc, 128..255 = speaker embedding
__global__ void k_cond(const float* __restrict__ lc, const float* __restrict__ embw,
                       const float* __restrict__ embb, const int* __restrict__ spk,
                       unsigned short* __restrict__ cond) {
  int n = blockIdx.z;
  long long idx = (long long)blockIdx.x * blockDim.x + threadIdx.x;
  if (idx >= 256ll * TT) return;
  int c = (int)(idx & 255), t = (int)(idx >> 8);
  float v = (c < 128) ? lc[((size_t)(n*128) + c)*TT + t]
                      : embw[(size_t)(c-128)*40 + spk[n]] + embb[c-128];
  cond[((size_t)n*SP + t)*256 + c] = f2bf(v);
}

// wav f32 (channel-major) -> bf16 time-major [n][t][256]
__global__ void k_towav(const float* __restrict__ wav, unsigned short* __restrict__ wb) {
  int n = blockIdx.z;
  long long idx = (long long)blockIdx.x * blockDim.x + threadIdx.x;
  if (idx >= 256ll * TT) return;
  int q = (int)(idx & 255), t = (int)(idx >> 8);
  wb[((size_t)n*SP + t)*256 + q] = f2bf(wav[((size_t)(n*256) + q)*TT + t]);
}

// skip f32 time-major -> relu -> h1 bf16 time-major
__global__ void k_reluskip(const float* __restrict__ skip, unsigned short* __restrict__ h1) {
  int n = blockIdx.z;
  long long idx = (long long)blockIdx.x * blockDim.x + threadIdx.x;
  if (idx >= 256ll * TOUT) return;
  int c = (int)(idx & 255), t = (int)(idx >> 8);
  float v = skip[((size_t)n*TOUTP + t)*256 + c];
  h1[((size_t)n*TOUTP + t)*256 + c] = f2bf(v > 0.f ? v : 0.f);
}

// ---------------- fused gated WMMA layer kernel ----------------
// filt/gate = W(256x1024) x Xcat(1024 x Lout); z = tanh(filt)*sigmoid(gate)
// activations time-major; K segments: [0,384) x[t] | [384,768) x[t+dil] | [768,1024) cond[cl+t]
__global__ __launch_bounds__(256) void k_gated(
    const unsigned short* __restrict__ Wf, const unsigned short* __restrict__ Wg,
    const float* __restrict__ bsig, const float* __restrict__ bgate,
    const unsigned short* __restrict__ X,   // [n][SP][384]
    const unsigned short* __restrict__ C,   // [n][SP][256]
    unsigned short* __restrict__ Z,         // [n][SP][256]
    int Lout, int dil, int cl) {
  __shared__ unsigned short sB[2 * 64 * 40];   // double-buffered [t][k], padded stride 40
  const int n = blockIdx.z;
  const int dblk = blockIdx.y * 64;
  const int tbase = blockIdx.x * 64;
  const int tid = threadIdx.x;
  const int lane = tid & 31;
  const int w = tid >> 5;
  const int wd = w & 3;                        // d-tile 0..3
  const int wh = w >> 2;                       // t-half 0..1
  const int m = lane & 15;
  const int kb = (lane >> 4) << 3;             // 0 or 8

  // staging: one aligned 16B chunk per thread (time-major rows => always aligned)
  const int tl = tid & 63;
  const int kc = tid >> 6;                     // 0..3: which 8-wide k chunk of 32
  const int tr = tbase + tl;
  const int trow = (tr < Lout) ? tr : (Lout - 1);   // clamp: garbage rows never consumed
  const unsigned short* p1 = X + ((size_t)n*SP + trow) * 384 + kc * 8;
  const unsigned short* p2 = X + ((size_t)n*SP + trow + dil) * 384 + kc * 8;
  const unsigned short* pc = C + ((size_t)n*SP + cl + trow) * 256 + kc * 8;
  unsigned short* sp = &sB[tl * 40 + kc * 8];

  const unsigned short* wf = Wf + (size_t)(dblk + wd*16 + m) * 1024 + kb;
  const unsigned short* wg = Wg + (size_t)(dblk + wd*16 + m) * 1024 + kb;
  const unsigned short* r0 = &sB[(wh*32 + m) * 40 + kb];
  const unsigned short* r1 = &sB[(wh*32 + 16 + m) * 40 + kb];

  f32x8 accf0 = {}, accf1 = {}, accg0 = {}, accg1 = {};

  async_ld_b128(sp, p1);                       // prologue: stage kt=0 into buffer 0
  for (int kt = 0; kt < 1024; kt += 32) {
    const int curoff = ((kt >> 5) & 1) * (64 * 40);
    wait_async0();                             // staged tile for this iteration landed
    __syncthreads();
    if (kt + 32 < 1024) {                      // stage next tile into other buffer (async)
      int kn = kt + 32;
      const unsigned short* g = (kn < 384) ? (p1 + kn)
                              : (kn < 768) ? (p2 + (kn - 384))
                                           : (pc + (kn - 768));
      async_ld_b128(sp + ((64*40) - curoff), g);
      __builtin_prefetch(wf + kn, 0, 3);       // -> global_prefetch_b8
      __builtin_prefetch(wg + kn, 0, 3);
    }
    union Frag { bf16x16 v; uint4 q[2]; } af, ag, b0, b1;
    af.q[0] = *(const uint4*)(wf + kt);        af.q[1] = *(const uint4*)(wf + kt + 16);
    ag.q[0] = *(const uint4*)(wg + kt);        ag.q[1] = *(const uint4*)(wg + kt + 16);
    b0.q[0] = *(const uint4*)(r0 + curoff);    b0.q[1] = *(const uint4*)(r0 + curoff + 16);
    b1.q[0] = *(const uint4*)(r1 + curoff);    b1.q[1] = *(const uint4*)(r1 + curoff + 16);
    accf0 = WMMA_BF16(af.v, b0.v, accf0);
    accf1 = WMMA_BF16(af.v, b1.v, accf1);
    accg0 = WMMA_BF16(ag.v, b0.v, accg0);
    accg1 = WMMA_BF16(ag.v, b1.v, accg1);
  }

  const int drow0 = dblk + wd*16 + ((lane >> 4) << 3);
#pragma unroll
  for (int h = 0; h < 2; ++h) {
    int tg = tbase + wh*32 + h*16 + m;
    if (tg >= Lout) continue;
    f32x8 f, g;
    if (h) { f = accf1; g = accg1; } else { f = accf0; g = accg0; }
    union { uint4 q; unsigned short s[8]; } op;
#pragma unroll
    for (int v = 0; v < 8; ++v) {
      float ff = f[v] + bsig[drow0 + v];
      float gg = g[v] + bgate[drow0 + v];
      op.s[v] = f2bf(tanhf(ff) * (1.0f / (1.0f + __expf(-gg))));
    }
    *(uint4*)(Z + ((size_t)n*SP + tg) * 256 + drow0) = op.q;   // packed b128 store
  }
}

// ---------------- generic 1x1-conv WMMA GEMM (time-major activations) ----------------
// Out(O x Tlen) = W(O x K) x In  [+ bias][+ bf16 resid][+ f32 acc][relu]
__global__ __launch_bounds__(256) void k_gemm(
    const unsigned short* __restrict__ W,     // O x K row-major
    const unsigned short* __restrict__ In,    // [n][in_rows][K]
    int in_rows, int in_t_off,
    int K, int O, int Tlen,
    const float* __restrict__ bias,
    const unsigned short* __restrict__ resid, int res_rows, int res_t_off,  // [n][res_rows][O]
    const float* facc, int facc_rows,                                        // [n][facc_rows][O]
    unsigned short* outb, int outb_rows,                                     // [n][outb_rows][O]
    float* outf, int outf_rows,                                              // [n][outf_rows][O]
    int relu) {
  __shared__ unsigned short sB[2 * 64 * 40];
  const int n = blockIdx.z;
  const int oblk = blockIdx.y * 64;
  const int tbase = blockIdx.x * 64;
  const int tid = threadIdx.x;
  const int lane = tid & 31;
  const int w = tid >> 5;
  const int wd = w & 3;
  const int wh = w >> 2;
  const int m = lane & 15;
  const int kb = (lane >> 4) << 3;

  const int tl = tid & 63;
  const int kc = tid >> 6;
  const int tr = tbase + tl;
  const int trow = (tr < Tlen) ? tr : (Tlen - 1);
  const unsigned short* p = In + ((size_t)n*in_rows + in_t_off + trow) * K + kc * 8;
  unsigned short* sp = &sB[tl * 40 + kc * 8];

  const unsigned short* wp = W + (size_t)(oblk + wd*16 + m) * K + kb;
  const unsigned short* r0 = &sB[(wh*32 + m) * 40 + kb];
  const unsigned short* r1 = &sB[(wh*32 + 16 + m) * 40 + kb];

  f32x8 acc0 = {}, acc1 = {};

  async_ld_b128(sp, p);
  for (int kt = 0; kt < K; kt += 32) {
    const int curoff = ((kt >> 5) & 1) * (64 * 40);
    wait_async0();
    __syncthreads();
    if (kt + 32 < K) {
      async_ld_b128(sp + ((64*40) - curoff), p + kt + 32);
      __builtin_prefetch(wp + kt + 32, 0, 3);
    }
    union Frag { bf16x16 v; uint4 q[2]; } a, b0, b1;
    a.q[0]  = *(const uint4*)(wp + kt);       a.q[1]  = *(const uint4*)(wp + kt + 16);
    b0.q[0] = *(const uint4*)(r0 + curoff);   b0.q[1] = *(const uint4*)(r0 + curoff + 16);
    b1.q[0] = *(const uint4*)(r1 + curoff);   b1.q[1] = *(const uint4*)(r1 + curoff + 16);
    acc0 = WMMA_BF16(a.v, b0.v, acc0);
    acc1 = WMMA_BF16(a.v, b1.v, acc1);
  }

  const int orow0 = oblk + wd*16 + ((lane >> 4) << 3);
#pragma unroll
  for (int h = 0; h < 2; ++h) {
    int tg = tbase + wh*32 + h*16 + m;
    if (tg >= Tlen) continue;
    f32x8 a = h ? acc1 : acc0;
    float val[8];
#pragma unroll
    for (int v = 0; v < 8; ++v) val[v] = a[v];
    if (bias) {
#pragma unroll
      for (int v = 0; v < 8; ++v) val[v] += bias[orow0 + v];
    }
    if (resid) {
      union { uint4 q; unsigned short s[8]; } rp;
      rp.q = *(const uint4*)(resid + ((size_t)n*res_rows + res_t_off + tg) * O + orow0);
#pragma unroll
      for (int v = 0; v < 8; ++v) val[v] += bf2f(rp.s[v]);
    }
    if (facc) {
      const float* fp = facc + ((size_t)n*facc_rows + tg) * O + orow0;
#pragma unroll
      for (int v = 0; v < 8; ++v) val[v] += fp[v];
    }
    if (relu) {
#pragma unroll
      for (int v = 0; v < 8; ++v) val[v] = val[v] > 0.f ? val[v] : 0.f;
    }
    if (outb) {
      union { uint4 q; unsigned short s[8]; } op;
#pragma unroll
      for (int v = 0; v < 8; ++v) op.s[v] = f2bf(val[v]);
      *(uint4*)(outb + ((size_t)n*outb_rows + tg) * O + orow0) = op.q;
    }
    if (outf) {
      float* fo = outf + ((size_t)n*outf_rows + tg) * O + orow0;
#pragma unroll
      for (int v = 0; v < 8; ++v) fo[v] = val[v];
    }
  }
}

// ---------------- log-softmax: read time-major logits, write channel-major d_out ----------------
__global__ __launch_bounds__(256) void k_lsm(const float* __restrict__ ltm, float* __restrict__ out) {
  __shared__ float red[256];
  int t = blockIdx.x, n = blockIdx.y, c = threadIdx.x;
  float v = ltm[((size_t)n*TOUTP + t) * 256 + c];          // coalesced read
  red[c] = v; __syncthreads();
  for (int s = 128; s > 0; s >>= 1) { if (c < s) red[c] = fmaxf(red[c], red[c+s]); __syncthreads(); }
  float mx = red[0]; __syncthreads();
  red[c] = __expf(v - mx); __syncthreads();
  for (int s = 128; s > 0; s >>= 1) { if (c < s) red[c] += red[c+s]; __syncthreads(); }
  float lse = mx + logf(red[0]);
  out[((size_t)n*256 + c) * TOUT + t] = v - lse;
}

// ---------------- host ----------------
static inline unsigned cdiv(long long a, long long b) { return (unsigned)((a + b - 1) / b); }

extern "C" void kernel_launch(void* const* d_in, const int* in_sizes, int n_in,
                              void* d_out, int out_size, void* d_ws, size_t ws_size,
                              hipStream_t stream) {
  (void)in_sizes; (void)n_in; (void)out_size; (void)ws_size;
  const float* wav       = (const float*)d_in[0];
  const float* lc_sparse = (const float*)d_in[1];
  const float* lc_conv_w = (const float*)d_in[2];
  const float* lc_conv_b = (const float*)d_in[3];
  const float* ups_w[4]  = {(const float*)d_in[4], (const float*)d_in[6], (const float*)d_in[8], (const float*)d_in[10]};
  const float* ups_b[4]  = {(const float*)d_in[5], (const float*)d_in[7], (const float*)d_in[9], (const float*)d_in[11]};
  const float* emb_w     = (const float*)d_in[12];
  const float* emb_b     = (const float*)d_in[13];
  const float* base_w    = (const float*)d_in[14];
  const float* base_b    = (const float*)d_in[15];
  const float* sig_w     = (const float*)d_in[16];
  const float* sig_b     = (const float*)d_in[17];
  const float* gate_w    = (const float*)d_in[18];
  const float* gate_b    = (const float*)d_in[19];
  const float* psig_w    = (const float*)d_in[20];
  const float* pgate_w   = (const float*)d_in[21];
  const float* skp_w     = (const float*)d_in[22];
  const float* res_w     = (const float*)d_in[23];
  const float* post1_w   = (const float*)d_in[24];
  const float* post1_b   = (const float*)d_in[25];
  const float* post2_w   = (const float*)d_in[26];
  const float* post2_b   = (const float*)d_in[27];
  const int*   spk       = (const int*)d_in[28];
  const int*   jit       = (const int*)d_in[29];

  char* ws = (char*)d_ws;
  size_t off = 0;
  auto alloc = [&](size_t bytes) -> char* {
    char* p = ws + off; off = (off + bytes + 255) & ~(size_t)255; return p;
  };
  unsigned short* WF   = (unsigned short*)alloc(20ull*256*1024*2);
  unsigned short* WG   = (unsigned short*)alloc(20ull*256*1024*2);
  unsigned short* WS   = (unsigned short*)alloc(20ull*256*256*2);
  unsigned short* WR   = (unsigned short*)alloc(20ull*384*256*2);
  unsigned short* WB   = (unsigned short*)alloc(384ull*256*2);
  unsigned short* WP1  = (unsigned short*)alloc(512ull*256*2);
  unsigned short* WP2  = (unsigned short*)alloc(256ull*512*2);
  unsigned short* WAVB = (unsigned short*)alloc(4ull*SP*256*2);   // later reused as H1
  unsigned short* COND = (unsigned short*)alloc(4ull*SP*256*2);
  unsigned short* XA   = (unsigned short*)alloc(4ull*SP*384*2);   // f32 lc scratch aliases here too
  unsigned short* XB   = (unsigned short*)alloc(4ull*SP*384*2);
  unsigned short* ZB   = (unsigned short*)alloc(4ull*SP*256*2);
  float*          SKIP = (float*)alloc(4ull*TOUTP*256*4);         // later reused as time-major logits
  unsigned short* H2   = (unsigned short*)alloc(4ull*TOUTP*512*2);
  unsigned short* H1   = WAVB;  // reuse: wav bf16 no longer needed after base conv
  float* fxA = (float*)XA;
  float* fxB = (float*)XB;

  // --- weight prepack (bf16) ---
  k_pack_fg<<<cdiv(20ll*256*1024, 256), 256, 0, stream>>>(sig_w, gate_w, psig_w, pgate_w, WF, WG);
  k_tob<<<cdiv(20ll*256*256, 256), 256, 0, stream>>>(skp_w,  WS,  20ll*256*256);
  k_tob<<<cdiv(20ll*384*256, 256), 256, 0, stream>>>(res_w,  WR,  20ll*384*256);
  k_tob<<<cdiv(384ll*256,   256), 256, 0, stream>>>(base_w,  WB,  384ll*256);
  k_tob<<<cdiv(512ll*256,   256), 256, 0, stream>>>(post1_w, WP1, 512ll*256);
  k_tob<<<cdiv(256ll*512,   256), 256, 0, stream>>>(post2_w, WP2, 256ll*512);

  // --- conditioning pipeline (lengths: 62 -> 290 -> 1148 -> 4580 -> 18308) ---
  k_lc0<<<dim3(cdiv(128ll*62, 256), 1, 4), 256, 0, stream>>>(lc_sparse, jit, lc_conv_w, lc_conv_b, fxA);
  k_tconv<<<dim3(cdiv(128ll*290,   256), 1, 4), 256, 0, stream>>>(fxA, ups_w[0], ups_b[0], fxB, 128, 62,   290,   25, 5);
  k_tconv<<<dim3(cdiv(128ll*1148,  256), 1, 4), 256, 0, stream>>>(fxB, ups_w[1], ups_b[1], fxA, 128, 290,  1148,  16, 4);
  k_tconv<<<dim3(cdiv(128ll*4580,  256), 1, 4), 256, 0, stream>>>(fxA, ups_w[2], ups_b[2], fxB, 128, 1148, 4580,  16, 4);
  k_tconv<<<dim3(cdiv(128ll*18308, 256), 1, 4), 256, 0, stream>>>(fxB, ups_w[3], ups_b[3], fxA, 128, 4580, 18308, 16, 4);
  k_cond<<<dim3(cdiv(256ll*TT, 256), 1, 4), 256, 0, stream>>>(fxA, emb_w, emb_b, spk, COND);

  // --- wav -> bf16 (time-major), base conv (x0 into XB), zero skip ---
  k_towav<<<dim3(cdiv(256ll*TT, 256), 1, 4), 256, 0, stream>>>(wav, WAVB);
  k_gemm<<<dim3(cdiv(TT, 64), 6, 4), 256, 0, stream>>>(
      WB, WAVB, SP, 0, 256, 384, TT, base_b,
      nullptr, 0, 0, nullptr, 0, XB, SP, nullptr, 0, 0);
  k_zero<<<cdiv(4ll*TOUTP*256, 256), 256, 0, stream>>>(SKIP, 4ll*TOUTP*256);

  // --- 20 gated residual layers ---
  const int DILS[NLAYER] = {1,2,4,8,16,32,64,128,256,512, 1,2,4,8,16,32,64,128,256,512};
  unsigned short* xcur = XB;
  unsigned short* xnxt = XA;
  int Lcur = TT, cum = 0;
  for (int i = 0; i < NLAYER; ++i) {
    int dil = DILS[i];
    int Lout = Lcur - dil;
    int cl = cum + dil;
    int sl = 2046 - cl;
    k_gated<<<dim3(cdiv(Lout, 64), 4, 4), 256, 0, stream>>>(
        WF + (size_t)i*256*1024, WG + (size_t)i*256*1024,
        sig_b + i*256, gate_b + i*256, xcur, COND, ZB, Lout, dil, cl);
    // skip: SKIP += skp_w * z[:, :, sl:]
    k_gemm<<<dim3(cdiv(TOUT, 64), 4, 4), 256, 0, stream>>>(
        WS + (size_t)i*256*256, ZB, SP, sl, 256, 256, TOUT, nullptr,
        nullptr, 0, 0, SKIP, TOUTP, nullptr, 0, SKIP, TOUTP, 0);
    // residual: x_next = res_w * z + x_cur[dil:]
    k_gemm<<<dim3(cdiv(Lout, 64), 6, 4), 256, 0, stream>>>(
        WR + (size_t)i*384*256, ZB, SP, 0, 256, 384, Lout, nullptr,
        xcur, SP, dil, nullptr, 0, xnxt, SP, nullptr, 0, 0);
    unsigned short* tmp = xcur; xcur = xnxt; xnxt = tmp;
    cum = cl; Lcur = Lout;
  }

  // --- post network ---
  k_reluskip<<<dim3(cdiv(256ll*TOUT, 256), 1, 4), 256, 0, stream>>>(SKIP, H1);
  k_gemm<<<dim3(cdiv(TOUT, 64), 8, 4), 256, 0, stream>>>(
      WP1, H1, TOUTP, 0, 256, 512, TOUT, post1_b,
      nullptr, 0, 0, nullptr, 0, H2, TOUTP, nullptr, 0, 1);
  k_gemm<<<dim3(cdiv(TOUT, 64), 4, 4), 256, 0, stream>>>(
      WP2, H2, TOUTP, 0, 512, 256, TOUT, post2_b,
      nullptr, 0, 0, nullptr, 0, nullptr, 0, (float*)SKIP, TOUTP, 0);
  k_lsm<<<dim3(TOUT, 4), 256, 0, stream>>>((const float*)SKIP, (float*)d_out);
}
